// EdgeBlock_2439541424687
// MI455X (gfx1250) — matrix operand order repaired and checked by reading, not compiled, not run on hardware
//
#include <hip/hip_runtime.h>

typedef _Float16 f16;
typedef __attribute__((ext_vector_type(16))) _Float16 v16h;
typedef __attribute__((ext_vector_type(8)))  _Float16 v8h;
typedef __attribute__((ext_vector_type(8)))  float    v8f;

#define B_   8
#define CIN  128
#define NPTS 2048
#define KNN  20
#define FM   128
#define FO   256

// ---- workspace layout (bytes, 256-aligned). Total ~4.03 MB ----
#define OFF_W1H 0u          // f16 [128][128]
#define OFF_W2H 32768u      // f16 [256][128]
#define OFF_WXH 98304u      // f16 [256][256]
#define OFF_WOH 229376u     // f16 [256][5120]  (f = K*256 + c)
#define OFF_B1F 2850816u    // f32 [128]
#define OFF_B2F 2851328u    // f32 [256]
#define OFF_BXF 2852352u    // f32 [256]
#define OFF_SQ  2853376u    // f32 [8][2048]
#define OFF_IDX 2918912u    // i32 [8][2048][20]

// fused-kernel dynamic LDS: idx 1280 + cen 4096 + diff 81920 + hx 163840 + h1 32768
#define SMEM_BYTES 283904

__device__ __forceinline__ float lrelu(float v){ return v >= 0.f ? v : 0.01f*v; }

__device__ __forceinline__ v16h comb(v8h lo, v8h hi){
  v16h r;
#pragma unroll
  for (int i=0;i<8;++i){ r[i]=lo[i]; r[i+8]=hi[i]; }
  return r;
}
// A fragment (16x32 f16): lanes0-15 K 0..15, lanes16-31 K 16..31
__device__ __forceinline__ v16h load_a16(const f16* __restrict__ row, int kbase, int lhi){
  const f16* p = row + kbase + lhi*16;
  return comb(*(const v8h*)p, *(const v8h*)(p+8));
}
// B fragment (32x16 f16): caller passes &feat[col][kbase + lhi*16]; 16 contiguous K
__device__ __forceinline__ v16h load_b16(const f16* p){
  return comb(*(const v8h*)p, *(const v8h*)(p+8));
}
__device__ __forceinline__ v8f wmma16(v16h a, v16h b, v8f c){
  return __builtin_amdgcn_wmma_f32_16x16x32_f16(false, a, false, b, (short)0, c, false, false);
}

// ---------------- prep: fold BN into weights, convert to f16 ----------------
__global__ __launch_bounds__(256) void prep_kernel(
    const float* __restrict__ w1, const float* __restrict__ b1, const float* __restrict__ g1,
    const float* __restrict__ be1, const float* __restrict__ m1, const float* __restrict__ v1,
    const float* __restrict__ w2, const float* __restrict__ b2, const float* __restrict__ g2,
    const float* __restrict__ be2, const float* __restrict__ m2, const float* __restrict__ v2,
    const float* __restrict__ wx, const float* __restrict__ bx, const float* __restrict__ gx,
    const float* __restrict__ bex, const float* __restrict__ mx, const float* __restrict__ vx,
    const float* __restrict__ wo,
    f16* __restrict__ w1h, f16* __restrict__ w2h, f16* __restrict__ wxh, f16* __restrict__ woh,
    float* __restrict__ b1f, float* __restrict__ b2f, float* __restrict__ bxf)
{
  int tid = blockIdx.x*256 + threadIdx.x;
  int stride = gridDim.x*256;
  for (int i = tid; i < FM*CIN; i += stride){
    int o = i >> 7; float inv = g1[o]*rsqrtf(v1[o]+1e-5f);
    w1h[i] = (f16)(w1[i]*inv);
  }
  for (int i = tid; i < FO*FM; i += stride){
    int o = i >> 7; float inv = g2[o]*rsqrtf(v2[o]+1e-5f);
    w2h[i] = (f16)(w2[i]*inv);
  }
  for (int i = tid; i < FO*2*CIN; i += stride){
    int o = i >> 8; float inv = gx[o]*rsqrtf(vx[o]+1e-5f);
    wxh[i] = (f16)(wx[i]*inv);
  }
  for (int i = tid; i < FO*FO*KNN; i += stride){
    int o = i / (FO*KNN); int f = i - o*(FO*KNN);
    int K = f >> 8; int c = f & 255;               // f = K*256 + c
    woh[i] = (f16)wo[((size_t)o*FO + c)*KNN + K];
  }
  for (int i = tid; i < FM; i += stride){
    float inv = g1[i]*rsqrtf(v1[i]+1e-5f);
    b1f[i] = b1[i]*inv + be1[i] - m1[i]*inv;
  }
  for (int i = tid; i < FO; i += stride){
    float inv = g2[i]*rsqrtf(v2[i]+1e-5f);
    b2f[i] = b2[i]*inv + be2[i] - m2[i]*inv;
  }
  for (int i = tid; i < FO; i += stride){
    float inv = gx[i]*rsqrtf(vx[i]+1e-5f);
    bxf[i] = bx[i]*inv + bex[i] - mx[i]*inv;
  }
}

// ---------------- squared norms ----------------
__global__ __launch_bounds__(256) void sq_kernel(const float* __restrict__ x, float* __restrict__ sq){
  int q = blockIdx.x*256 + threadIdx.x;
  if (q >= B_*NPTS) return;
  int b = q >> 11, n = q & (NPTS-1);
  float s = 0.f;
#pragma unroll 8
  for (int c = 0; c < CIN; ++c){
    float v = x[((size_t)b*CIN + c)*NPTS + n];
    s = fmaf(v, v, s);
  }
  sq[q] = s;
}

// ---------------- kNN: one wave per query, register top-20 + 32-way merge ----------------
__global__ __launch_bounds__(256) void knn_kernel(const float* __restrict__ x,
                                                  const float* __restrict__ sq,
                                                  int* __restrict__ gidx){
  __shared__ float cen[8*CIN];
  __shared__ float mkey[8*32*KNN];
  __shared__ int   mid [8*32*KNN];
  const int b  = blockIdx.x >> 8;          // 256 blocks per batch (8 queries/block)
  const int ib = blockIdx.x & 255;
  const int t = threadIdx.x, wid = t >> 5, lane = t & 31;

  for (int q = t; q < 8*CIN; q += 256){
    int w = q >> 7, c = q & 127;
    cen[q] = x[((size_t)b*CIN + c)*NPTS + (ib*8 + w)];
  }
  __syncthreads();

  const int i = ib*8 + wid;
  float d[KNN]; int id[KNN];
#pragma unroll
  for (int s = 0; s < KNN; ++s){ d[s] = 3.4e38f; id[s] = 0; }
  const float* cw = cen + wid*CIN;

  for (int tj = 0; tj < NPTS/32; ++tj){
    int j = tj*32 + lane;
    float dot = 0.f;
#pragma unroll 8
    for (int c = 0; c < CIN; ++c)
      dot = fmaf(x[((size_t)b*CIN + c)*NPTS + j], cw[c], dot);
    float key = sq[b*NPTS + j] - 2.f*dot;   // + ||xi||^2 is rank-invariant
    if (j != i){
      float kd = key; int ki = j;
#pragma unroll
      for (int s = 0; s < KNN; ++s){
        if (kd < d[s]){ float td=d[s]; int ti=id[s]; d[s]=kd; id[s]=ki; kd=td; ki=ti; }
      }
    }
  }
#pragma unroll
  for (int s = 0; s < KNN; ++s){
    mkey[(wid*32+lane)*KNN + s] = d[s];
    mid [(wid*32+lane)*KNN + s] = id[s];
  }
  __syncthreads();
  if (lane == 0){
    int head[32];
    for (int l = 0; l < 32; ++l) head[l] = 0;
    int* outp = gidx + ((size_t)b*NPTS + i)*KNN;
    for (int r = 0; r < KNN; ++r){
      float best = 3.5e38f; int bl = 0;
      for (int l = 0; l < 32; ++l){
        float v = mkey[(wid*32+l)*KNN + head[l]];
        if (v < best){ best = v; bl = l; }
      }
      outp[r] = mid[(wid*32+bl)*KNN + head[bl]];
      head[bl]++;
    }
  }
}

// ---------------- fused edge block: gather -> GEMM1 -> GEMM2 -> softmax ->
// ---------------- GEMM3*softmax -> final wo contraction, all in LDS ----------------
__global__ __launch_bounds__(256) void fused_kernel(
    const float* __restrict__ x, const int* __restrict__ gidx,
    const f16* __restrict__ w1h, const float* __restrict__ b1f,
    const f16* __restrict__ w2h, const float* __restrict__ b2f,
    const f16* __restrict__ wxh, const float* __restrict__ bxf,
    const f16* __restrict__ woh, const float* __restrict__ bo,
    float* __restrict__ out)
{
  extern __shared__ __align__(128) char smem[];
  int* sIdx  = (int*)smem;                               // [320]
  f16* sCen  = (f16*)(smem + 1280);                      // [16][128]
  f16* sDiff = (f16*)(smem + 1280 + 4096);               // [320][128]
  f16* sHx   = (f16*)(smem + 1280 + 4096 + 81920);       // [320][256] h2 -> softmax -> xf
  f16* sH1   = (f16*)(smem + 1280 + 4096 + 81920 + 163840); // [8 waves][16][128]

  const int b   = blockIdx.x >> 7;         // 128 point-tiles per batch
  const int n0  = (blockIdx.x & 127) * 16;
  const int t   = threadIdx.x;
  const int wid = t >> 5;
  const int lane = t & 31;
  const int l16 = lane & 15;
  const int lhi = lane >> 4;

  // ---- stage idx + central ----
  for (int q = t; q < 16*KNN; q += 256)
    sIdx[q] = gidx[((size_t)(b*NPTS + n0 + q/KNN))*KNN + (q % KNN)];
  for (int q = t; q < 16*CIN; q += 256){
    int nl = q >> 7, c = q & 127;
    sCen[q] = (f16)x[((size_t)b*CIN + c)*NPTS + (n0 + nl)];
  }
  __syncthreads();
  // ---- diff = x[:,j] - x[:,n] ----
  for (int q = t; q < 320*CIN; q += 256){
    int e = q >> 7, c = q & 127;
    int nl = e / KNN;
    int j = sIdx[e];
    float xj = x[((size_t)b*CIN + c)*NPTS + j];
    sDiff[q] = (f16)(xj - (float)sCen[nl*CIN + c]);
  }
  __syncthreads();

  // ---- GEMM1 (w1 @ diff) fused with GEMM2 (w2 @ h1), 20 edge-tiles over 8 waves ----
  for (int r = 0; r < 3; ++r){
    int et = r*8 + wid;
    bool act = (et < 20);
    if (act){
      f16* h1 = sH1 + wid*(16*FM);
      for (int mo = 0; mo < FM/16; ++mo){
        v8f acc = {};
#pragma unroll
        for (int kc = 0; kc < CIN/32; ++kc){
          v16h a  = load_a16(w1h + (mo*16 + l16)*CIN, kc*32, lhi);
          v16h bb = load_b16(sDiff + (et*16 + l16)*CIN + kc*32 + lhi*16);
          acc = wmma16(a, bb, acc);
        }
#pragma unroll
        for (int rr = 0; rr < 8; ++rr){
          int o = mo*16 + lhi*8 + rr;
          h1[l16*FM + o] = (f16)lrelu(acc[rr] + b1f[o]);
        }
      }
    }
    __syncthreads();   // cross-lane h1 visibility (uniform barrier)
    if (act){
      const f16* h1 = sH1 + wid*(16*FM);
      for (int mo = 0; mo < FO/16; ++mo){
        v8f acc = {};
#pragma unroll
        for (int kc = 0; kc < FM/32; ++kc){
          v16h a  = load_a16(w2h + (mo*16 + l16)*FM, kc*32, lhi);
          v16h bb = load_b16(h1 + l16*FM + kc*32 + lhi*16);
          acc = wmma16(a, bb, acc);
        }
        int e = et*16 + l16;
#pragma unroll
        for (int rr = 0; rr < 8; ++rr){
          int o = mo*16 + lhi*8 + rr;
          sHx[e*FO + o] = (f16)lrelu(acc[rr] + b2f[o]);
        }
      }
    }
    __syncthreads();
  }

  // ---- softmax over k for each (n,o) ----
  for (int p = t; p < 16*FO; p += 256){
    int nl = p >> 8, o = p & 255;
    f16* base = sHx + (nl*KNN)*FO + o;
    float mxv = -3.4e38f;
#pragma unroll
    for (int kk = 0; kk < KNN; ++kk) mxv = fmaxf(mxv, (float)base[kk*FO]);
    float ssum = 0.f;
#pragma unroll
    for (int kk = 0; kk < KNN; ++kk){
      float ev = __expf((float)base[kk*FO] - mxv);
      ssum += ev;
      base[kk*FO] = (f16)ev;
    }
    float inv = 1.f / ssum;
#pragma unroll
    for (int kk = 0; kk < KNN; ++kk)
      base[kk*FO] = (f16)((float)base[kk*FO] * inv);
  }
  __syncthreads();

  // ---- GEMM3: xf = lrelu(wx @ [cen;diff] + bx) * softmax, in-place into sHx ----
  for (int r = 0; r < 3; ++r){
    int et = r*8 + wid;
    if (et < 20){
      for (int mo = 0; mo < FO/16; ++mo){
        v8f acc = {};
        int e  = et*16 + l16;
        int nl = e / KNN;
#pragma unroll
        for (int kc = 0; kc < (2*CIN)/32; ++kc){
          v16h a = load_a16(wxh + (mo*16 + l16)*(2*CIN), kc*32, lhi);
          const f16* bp = (kc < 4)
              ? (sCen  + nl*CIN + kc*32 + lhi*16)
              : (sDiff + e*CIN  + (kc-4)*32 + lhi*16);
          v16h bb = load_b16(bp);
          acc = wmma16(a, bb, acc);
        }
#pragma unroll
        for (int rr = 0; rr < 8; ++rr){
          int o = mo*16 + lhi*8 + rr;
          float v = lrelu(acc[rr] + bxf[o]);
          float w = (float)sHx[e*FO + o];
          sHx[e*FO + o] = (f16)(v * w);
        }
      }
    }
  }
  __syncthreads();

  // ---- final: out[o][n] = sum_{K,c} woh[o][K*256+c] * xf[n][K][c] + bo[o] ----
  for (int mo = wid; mo < FO/16; mo += 8){
    v8f acc = {};
    const f16* arow = woh + (size_t)(mo*16 + l16) * (FO*KNN);
#pragma unroll 4
    for (int fc = 0; fc < (FO*KNN)/32; ++fc){
      v16h a = load_a16(arow, fc*32, lhi);
      int fbase = fc*32 + lhi*16;
      int K = fbase >> 8, c = fbase & 255;
      v16h bb = load_b16(sHx + (l16*KNN + K)*FO + c);
      acc = wmma16(a, bb, acc);
    }
#pragma unroll
    for (int rr = 0; rr < 8; ++rr){
      int o = mo*16 + lhi*8 + rr;
      out[((size_t)b*FO + o)*NPTS + (n0 + l16)] = acc[rr] + bo[o];
    }
  }
}

extern "C" void kernel_launch(void* const* d_in, const int* in_sizes, int n_in,
                              void* d_out, int out_size, void* d_ws, size_t ws_size,
                              hipStream_t stream) {
  (void)in_sizes; (void)n_in; (void)out_size; (void)ws_size;
  const float* x   = (const float*)d_in[0];
  const float* w1  = (const float*)d_in[1];
  const float* b1  = (const float*)d_in[2];
  const float* g1  = (const float*)d_in[3];
  const float* be1 = (const float*)d_in[4];
  const float* m1  = (const float*)d_in[5];
  const float* v1  = (const float*)d_in[6];
  const float* w2  = (const float*)d_in[7];
  const float* b2  = (const float*)d_in[8];
  const float* g2  = (const float*)d_in[9];
  const float* be2 = (const float*)d_in[10];
  const float* m2  = (const float*)d_in[11];
  const float* v2  = (const float*)d_in[12];
  const float* wx  = (const float*)d_in[13];
  const float* bx  = (const float*)d_in[14];
  const float* gx  = (const float*)d_in[15];
  const float* bex = (const float*)d_in[16];
  const float* mx  = (const float*)d_in[17];
  const float* vx  = (const float*)d_in[18];
  const float* wo  = (const float*)d_in[19];
  const float* bo  = (const float*)d_in[20];

  char* ws = (char*)d_ws;
  f16*   w1h = (f16*)(ws + OFF_W1H);
  f16*   w2h = (f16*)(ws + OFF_W2H);
  f16*   wxh = (f16*)(ws + OFF_WXH);
  f16*   woh = (f16*)(ws + OFF_WOH);
  float* b1f = (float*)(ws + OFF_B1F);
  float* b2f = (float*)(ws + OFF_B2F);
  float* bxf = (float*)(ws + OFF_BXF);
  float* sqb = (float*)(ws + OFF_SQ);
  int*   idx = (int*)(ws + OFF_IDX);

  prep_kernel<<<1024, 256, 0, stream>>>(w1,b1,g1,be1,m1,v1, w2,b2,g2,be2,m2,v2,
                                        wx,bx,gx,bex,mx,vx, wo,
                                        w1h,w2h,wxh,woh, b1f,b2f,bxf);
  sq_kernel<<<(B_*NPTS + 255)/256, 256, 0, stream>>>(x, sqb);
  knn_kernel<<<B_*NPTS/8, 256, 0, stream>>>(x, sqb, idx);
  fused_kernel<<<B_*NPTS/16, 256, SMEM_BYTES, stream>>>(x, idx,
                                                        w1h,b1f, w2h,b2f, wxh,bxf,
                                                        woh, bo, (float*)d_out);
}
